// GNNDenseBlock_36919538876773
// MI455X (gfx1250) — matrix operand order repaired and checked by reading, not compiled, not run on hardware
//
#include <hip/hip_runtime.h>
#include <hip/hip_bf16.h>

// ---------------------------------------------------------------------------
// GNN dense block for MI455X (gfx1250, wave32, WMMA).
//
// out = concat(x, h1, h2, h3), layer l:
//   h   = out[:, 0:dimIn)               (prefix of output buffer, stride 160)
//   agg = segment_sum(h[src], tgt)      (scatter kernel, f32 atomics -> L2)
//   y1  = agg@w_rel1 + h@w_root1 + b1   (dual-A WMMA bf16 GEMM, W in LDS)
//   h1  = relu(batchnorm(y1))
//   agg2= segment_sum(h1[src], tgt)
//   y2  = agg2@w_rel2 + h1@w_root2 + b2
//   out[:, 64+32l : 64+32(l+1)) = relu(batchnorm(y2))
//
// N_NODES = 50000 = 3125*16  ->  M is tile-aligned, no row guards anywhere.
// ---------------------------------------------------------------------------

#define N_NODES  50000
#define N_EDGES  800000
#define OUT_LD   160
#define BN_EPS   1e-5f

typedef __attribute__((ext_vector_type(16))) __bf16 v16bf;
typedef __attribute__((ext_vector_type(8)))  __bf16 v8bf;
typedef __attribute__((ext_vector_type(8)))  float  v8f;

union Frag16 { v16bf v; __bf16 e[16]; };
union Frag8  { v8f   v; float  e[8];  };

// ---------------------------------------------------------------------- utils
__global__ void zero_f32_kernel(float* __restrict__ p, long long n) {
  long long i = (long long)blockIdx.x * blockDim.x + threadIdx.x;
  if (i < n) p[i] = 0.0f;
}

__global__ void copy_x_kernel(const float* __restrict__ x, float* __restrict__ out) {
  long long i = (long long)blockIdx.x * blockDim.x + threadIdx.x;
  long long n = (long long)N_NODES * 64;
  if (i >= n) return;
  int r = (int)(i >> 6);          // /64
  int c = (int)(i & 63);
  out[(size_t)r * OUT_LD + c] = x[i];
}

// ------------------------------------------------------------------- scatter
// agg[tgt[e], :] += feat[src[e], :]; one thread per (edge, 4 features).
__global__ void scatter_add_kernel(const float* __restrict__ feat, int ldf,
                                   const int* __restrict__ src,
                                   const int* __restrict__ tgt,
                                   float* __restrict__ agg, int dim) {
  const int q = dim >> 2;                       // float4 groups per row
  long long tid = (long long)blockIdx.x * blockDim.x + threadIdx.x;
  long long total = (long long)N_EDGES * q;
  if (tid >= total) return;
  int e  = (int)(tid / q);
  int f4 = (int)(tid % q) << 2;
  int s = src[e], t = tgt[e];
  float4 v = *(const float4*)(feat + (size_t)s * ldf + f4);
  float* dst = agg + (size_t)t * dim + f4;
  atomicAdd(dst + 0, v.x);
  atomicAdd(dst + 1, v.y);
  atomicAdd(dst + 2, v.z);
  atomicAdd(dst + 3, v.w);
}

// ---------------------------------------------------------------- WMMA GEMM
// Y[:, colOff:colOff+Nc) = A1@W1 + A2@W2 + bias   (A: [M,K] f32, W: [K,Nc] f32)
// bf16 multiplicands, f32 accumulate via v_wmma_f32_16x16x32_bf16.
// W1/W2 are staged bf16-TRANSPOSED in LDS (WT[n*K + k]) so each lane's
// B-fragment is two runs of 8 contiguous bf16, 16 apart -> two ds_load_b128 per K-step.
// One wave owns a 16x16 output tile. M % 16 == 0 (50000 = 3125*16).
extern "C" __global__ void __launch_bounds__(256)
gemm_dual_wmma_kernel(const float* __restrict__ A1, int lda1,
                      const float* __restrict__ A2, int lda2,
                      const float* __restrict__ W1,
                      const float* __restrict__ W2, int ldw,
                      const float* __restrict__ bias,
                      float* __restrict__ Y, int ldy, int colOff,
                      int M, int K, int Nc) {
  extern __shared__ __bf16 ldsW[];              // 2 * K * Nc bf16

  // ---- stage W1,W2 transposed to bf16 LDS (all 256 threads cooperate)
  const int wsz = K * Nc;
  for (int idx = threadIdx.x; idx < 2 * wsz; idx += 256) {
    int which = idx >= wsz;
    int local = idx - which * wsz;
    int k = local / Nc;
    int n = local - k * Nc;
    const float* W = which ? W2 : W1;
    ldsW[which * wsz + n * K + k] = (__bf16)W[(size_t)k * ldw + n];
  }
  __syncthreads();

  const int lane   = threadIdx.x & 31;
  const int wave   = (int)(blockIdx.x * (blockDim.x >> 5) + (threadIdx.x >> 5));
  const int tilesN = Nc >> 4;
  const int tilesM = M >> 4;
  if (wave >= tilesM * tilesN) return;          // wave-uniform exit

  const int m0 = (wave / tilesN) << 4;
  const int n0 = (wave % tilesN) << 4;
  const int l15 = lane & 15;
  const bool hi = lane >= 16;
  const int mL = m0 + l15;                      // A-fragment row for this lane
  const int nL = n0 + l15;                      // B-fragment col for this lane

  v8f acc = {};

  #pragma unroll
  for (int pass = 0; pass < 2; ++pass) {
    const float*  A   = pass ? A2 : A1;
    const int     lda = pass ? lda2 : lda1;
    const __bf16* WT  = ldsW + pass * wsz + nL * K;
    for (int kb = 0; kb < K; kb += 32) {        // K is a multiple of 32 here
      // A 16x32 bf16 layout (ISA 7.12.2): lane group offset 0/8, elements
      // j=0..7 -> k contiguous, j=8..15 -> k+16 contiguous.
      const float* ap = A + (size_t)mL * lda + kb + (hi ? 8 : 0);
      float4 f0 = *(const float4*)(ap);
      float4 f1 = *(const float4*)(ap + 4);
      float4 f2 = *(const float4*)(ap + 16);
      float4 f3 = *(const float4*)(ap + 20);
      Frag16 a;
      a.e[0]  = (__bf16)f0.x; a.e[1]  = (__bf16)f0.y;
      a.e[2]  = (__bf16)f0.z; a.e[3]  = (__bf16)f0.w;
      a.e[4]  = (__bf16)f1.x; a.e[5]  = (__bf16)f1.y;
      a.e[6]  = (__bf16)f1.z; a.e[7]  = (__bf16)f1.w;
      a.e[8]  = (__bf16)f2.x; a.e[9]  = (__bf16)f2.y;
      a.e[10] = (__bf16)f2.z; a.e[11] = (__bf16)f2.w;
      a.e[12] = (__bf16)f3.x; a.e[13] = (__bf16)f3.y;
      a.e[14] = (__bf16)f3.z; a.e[15] = (__bf16)f3.w;

      // B 32x16 bf16 layout (matches A): lane group offset 0/8, elements
      // j=0..7 -> k contiguous, j=8..15 -> k+16 contiguous, col N = lane%16.
      const __bf16* wt = WT + kb + (hi ? 8 : 0);
      Frag16 b;
      *(v8bf*)&b.e[0] = *(const v8bf*)(wt);
      *(v8bf*)&b.e[8] = *(const v8bf*)(wt + 16);

      acc = __builtin_amdgcn_wmma_f32_16x16x32_bf16(
          /*neg_a=*/false, a.v, /*neg_b=*/false, b.v,
          /*c_mod=*/(short)0, acc, /*reuse_a=*/false, /*reuse_b=*/false);
    }
  }

  Frag8 c; c.v = acc;
  const float bn = bias[n0 + l15];
  #pragma unroll
  for (int r = 0; r < 8; ++r) {
    // C/D 16x16 f32 layout: VGPR r -> M = r (+8 for lanes 16-31), N = lane%16.
    int m = m0 + r + (hi ? 8 : 0);
    Y[(size_t)m * ldy + colOff + n0 + l15] = c.e[r] + bn;
  }
}

// --------------------------------------------------------------- batch norm
// stats[0:C) = colwise sum, stats[C:2C) = colwise sum of squares (pre-zeroed).
__global__ void bn_stats_kernel(const float* __restrict__ X, int ldx, int colOff,
                                int M, int C, float* __restrict__ stats) {
  const int c  = threadIdx.x % C;               // C is 32 or 128 (divides 256)
  const int rb = threadIdx.x / C;
  const int rowsPerBlock = blockDim.x / C;
  float s = 0.0f, q = 0.0f;
  for (int r = blockIdx.x * rowsPerBlock + rb; r < M;
       r += gridDim.x * rowsPerBlock) {
    float v = X[(size_t)r * ldx + colOff + c];
    s += v;
    q += v * v;
  }
  atomicAdd(&stats[c], s);
  atomicAdd(&stats[C + c], q);
}

__global__ void bn_apply_relu_kernel(float* __restrict__ X, int ldx, int colOff,
                                     int M, int C,
                                     const float* __restrict__ stats,
                                     const float* __restrict__ gamma,
                                     const float* __restrict__ beta) {
  long long i = (long long)blockIdx.x * blockDim.x + threadIdx.x;
  long long n = (long long)M * C;
  if (i >= n) return;
  int r = (int)(i / C);
  int c = (int)(i % C);
  float invM = 1.0f / (float)M;
  float mu  = stats[c] * invM;
  float var = stats[C + c] * invM - mu * mu;
  float v = X[(size_t)r * ldx + colOff + c];
  float y = gamma[c] * (v - mu) * rsqrtf(var + BN_EPS) + beta[c];
  X[(size_t)r * ldx + colOff + c] = y > 0.0f ? y : 0.0f;
}

// ------------------------------------------------------------------- launch
static inline void launch_zero(float* p, long long n, hipStream_t s) {
  long long blocks = (n + 255) / 256;
  zero_f32_kernel<<<(unsigned)blocks, 256, 0, s>>>(p, n);
}

extern "C" void kernel_launch(void* const* d_in, const int* in_sizes, int n_in,
                              void* d_out, int out_size, void* d_ws, size_t ws_size,
                              hipStream_t stream) {
  const float* x  = (const float*)d_in[0];
  const int* ei   = (const int*)d_in[1];
  const int* src  = ei;
  const int* tgt  = ei + N_EDGES;
  float* out      = (float*)d_out;

  // Workspace layout: agg [N,128] | h1 [N,128] | stats [256]
  float* agg   = (float*)d_ws;
  float* h1    = agg + (size_t)N_NODES * 128;
  float* stats = h1  + (size_t)N_NODES * 128;

  const int N = N_NODES;

  // out[:, 0:64) = x
  {
    long long n = (long long)N * 64;
    copy_x_kernel<<<(unsigned)((n + 255) / 256), 256, 0, stream>>>(x, out);
  }

  for (int l = 0; l < 3; ++l) {
    const float* P[10];
    for (int k = 0; k < 10; ++k) P[k] = (const float*)d_in[2 + l * 10 + k];
    const float* w_rel1  = P[0];
    const float* w_root1 = P[1];
    const float* b1      = P[2];
    const float* g1      = P[3];
    const float* beta1   = P[4];
    const float* w_rel2  = P[5];
    const float* w_root2 = P[6];
    const float* b2      = P[7];
    const float* g2      = P[8];
    const float* beta2   = P[9];

    const int dimIn  = 64 + 32 * l;   // 64, 96, 128
    const int dimMid = 128;
    const int dimOut = 32;
    const int colOff = 64 + 32 * l;

    // ---- conv1: agg = scatter(h), y1 = agg@w_rel1 + h@w_root1 + b1 -> h1
    launch_zero(agg, (long long)N * dimIn, stream);
    {
      long long tot = (long long)N_EDGES * (dimIn >> 2);
      scatter_add_kernel<<<(unsigned)((tot + 255) / 256), 256, 0, stream>>>(
          out, OUT_LD, src, tgt, agg, dimIn);
    }
    {
      int tiles = (N / 16) * (dimMid / 16);
      size_t shbytes = (size_t)2 * dimIn * dimMid * sizeof(__bf16);
      gemm_dual_wmma_kernel<<<(unsigned)((tiles + 7) / 8), 256, shbytes, stream>>>(
          agg, dimIn, out, OUT_LD, w_rel1, w_root1, dimMid, b1,
          h1, dimMid, 0, N, dimIn, dimMid);
    }
    launch_zero(stats, 2 * dimMid, stream);
    bn_stats_kernel<<<256, 256, 0, stream>>>(h1, dimMid, 0, N, dimMid, stats);
    {
      long long n = (long long)N * dimMid;
      bn_apply_relu_kernel<<<(unsigned)((n + 255) / 256), 256, 0, stream>>>(
          h1, dimMid, 0, N, dimMid, stats, g1, beta1);
    }

    // ---- conv2: agg2 = scatter(h1), y2 = agg2@w_rel2 + h1@w_root2 + b2
    launch_zero(agg, (long long)N * dimMid, stream);
    {
      long long tot = (long long)N_EDGES * (dimMid >> 2);
      scatter_add_kernel<<<(unsigned)((tot + 255) / 256), 256, 0, stream>>>(
          h1, dimMid, src, tgt, agg, dimMid);
    }
    {
      int tiles = (N / 16) * (dimOut / 16);
      size_t shbytes = (size_t)2 * dimMid * dimOut * sizeof(__bf16);
      gemm_dual_wmma_kernel<<<(unsigned)((tiles + 7) / 8), 256, shbytes, stream>>>(
          agg, dimMid, h1, dimMid, w_rel2, w_root2, dimOut, b2,
          out, OUT_LD, colOff, N, dimMid, dimOut);
    }
    launch_zero(stats, 2 * dimOut, stream);
    bn_stats_kernel<<<256, 256, 0, stream>>>(out, OUT_LD, colOff, N, dimOut, stats);
    {
      long long n = (long long)N * dimOut;
      bn_apply_relu_kernel<<<(unsigned)((n + 255) / 256), 256, 0, stream>>>(
          out, OUT_LD, colOff, N, dimOut, stats, g2, beta2);
    }
  }
}
